// SpikingLMUFFTCell_1082331758674
// MI455X (gfx1250) — matrix operand order repaired and checked by reading, not compiled, not run on hardware
//
#include <hip/hip_runtime.h>
#include <hip/hip_bf16.h>
#include <math.h>

// ---------------- problem constants ----------------
#define B_   16
#define S_   4096
#define I_   128
#define HD_  384
#define M_   256
#define K_   (M_ + I_)      // 384
#define NFFT 8192
#define LOGN 13
#define EPS_ 1e-5f

typedef __attribute__((ext_vector_type(16))) _Float16 v16h;
typedef __attribute__((ext_vector_type(8)))  float    v8f;
typedef unsigned int v4u __attribute__((ext_vector_type(4)));
typedef int          v8i __attribute__((ext_vector_type(8)));
typedef int          v4i __attribute__((ext_vector_type(4)));

struct alignas(16) H8 { _Float16 h[8]; };   // 16-byte packed half chunk

#if defined(__has_builtin)
#  if __has_builtin(__builtin_amdgcn_tensor_load_to_lds) && __has_builtin(__builtin_amdgcn_s_wait_tensorcnt)
#    define HAVE_TDM 1
#  else
#    define HAVE_TDM 0
#  endif
#else
#  define HAVE_TDM 0
#endif
// therock-10.0 headers ship the 6-arg builtin; ROCm 7.2 clang-22 has the 5-arg form
#if __has_include(<hip/amd_detail/amd_gfx1250_TDM.h>)
#  define TDM_6ARG 1
#else
#  define TDM_6ARG 0
#endif

// The TDM builtin addresses LDS by integer offset, so LLVM cannot see that it
// stores to smem[]. Escape the LDS pointer INTO the asm so the memory clobber
// can legally modify it; otherwise loads from the never-stored LDS global fold
// to undef (observed in round 2/3 disasm: wmma with A==B and no ds_loads).
__device__ __forceinline__ void lds_written_by_dma(const void* p) {
  asm volatile("" : : "v"(p) : "memory");
}

// ---------------- fp32 -> fp16 convert ----------------
__global__ void k_cvt_f2h(const float* __restrict__ src, _Float16* __restrict__ dst, int n) {
  int i = blockIdx.x * blockDim.x + threadIdx.x;
  if (i < n) dst[i] = (_Float16)src[i];
}

// ---------------- u branch: Linear(I->1) + BN ----------------
__global__ void __launch_bounds__(256)
k_upre(const float* __restrict__ x, const float* __restrict__ w,
       const float* __restrict__ wb, const float* __restrict__ g,
       const float* __restrict__ bb, const float* __restrict__ mu,
       const float* __restrict__ var, float* __restrict__ u) {
  int wv = threadIdx.x >> 5, lane = threadIdx.x & 31;
  int row = blockIdx.x * 8 + wv;                         // 0..B*S-1
  const float4* xr = (const float4*)(x + (size_t)row * I_);
  const float4* wr = (const float4*)w;
  float4 xv = xr[lane], wvv = wr[lane];
  float d = xv.x*wvv.x + xv.y*wvv.y + xv.z*wvv.z + xv.w*wvv.w;
  #pragma unroll
  for (int off = 16; off; off >>= 1) d += __shfl_down(d, off, 32);
  if (lane == 0) {
    float pre = d + wb[0];
    u[row] = g[0] * (pre - mu[0]) * rsqrtf(var[0] + EPS_) + bb[0];
  }
}

// ---------------- LIF scan for u (16 sequences, in place) ----------------
__global__ void k_ulif(float* __restrict__ u) {
  int b = threadIdx.x;
  if (b >= B_) return;
  float v = 0.f;
  for (int t = 0; t < S_; ++t) {
    size_t o = (size_t)b * S_ + t;
    v = 0.5f * (v + u[o]);                 // v += (x - v)/tau, tau=2
    float s = (v >= 1.0f) ? 1.f : 0.f;     // spike(v - v_th)
    u[o] = s;
    v *= (1.f - s);                        // hard reset
  }
}

// ---------------- radix-2 DIT FFT (8192 complex) in LDS ----------------
__device__ __forceinline__ void fft_lds(float2* s, int tid) {
  for (int i = tid; i < NFFT; i += 256) {
    int j = (int)(__brev((unsigned)i) >> (32 - LOGN));
    if (i < j) { float2 t = s[i]; s[i] = s[j]; s[j] = t; }
  }
  __syncthreads();
  for (int len = 2; len <= NFFT; len <<= 1) {
    int hl = len >> 1;
    float ang0 = -6.283185307179586f / (float)len;
    for (int k = tid; k < (NFFT >> 1); k += 256) {
      int grp = k / hl, pos = k - grp * hl;
      int i0 = grp * len + pos, i1 = i0 + hl;
      float sw, cw;
      __sincosf(ang0 * (float)pos, &sw, &cw);
      float2 a = s[i0], b = s[i1];
      float tr = b.x * cw - b.y * sw;
      float ti = b.x * sw + b.y * cw;
      s[i0] = make_float2(a.x + tr, a.y + ti);
      s[i1] = make_float2(a.x - tr, a.y - ti);
    }
    __syncthreads();
  }
}

__global__ void __launch_bounds__(256)
k_fft_fwd(const float* __restrict__ in, float2* __restrict__ out, int rowstride) {
  __shared__ float2 sbuf[NFFT];             // 64 KB of the 320 KB WGP LDS
  int row = blockIdx.x, tid = threadIdx.x;
  const float* src = in + (size_t)row * rowstride;
  for (int t = tid; t < NFFT; t += 256)
    sbuf[t] = make_float2(t < S_ ? src[t] : 0.f, 0.f);
  __syncthreads();
  fft_lds(sbuf, tid);
  float2* dst = out + (size_t)row * NFFT;
  for (int f = tid; f < NFFT; f += 256) dst[f] = sbuf[f];
}

// pointwise spectral product + inverse FFT (conj trick) + BN -> m_bn[b][ch][t] fp16
__global__ void __launch_bounds__(256)
k_conv_ifft(const float2* __restrict__ Uf, const float2* __restrict__ Hf,
            const float* __restrict__ g, const float* __restrict__ bb,
            const float* __restrict__ mu, const float* __restrict__ var,
            _Float16* __restrict__ mbn) {
  __shared__ float2 sbuf[NFFT];
  int b = blockIdx.x >> 8, ch = blockIdx.x & (M_ - 1), tid = threadIdx.x;
  const float2* uf = Uf + (size_t)b * NFFT;
  const float2* hf = Hf + (size_t)ch * NFFT;
  for (int f = tid; f < NFFT; f += 256) {
    float2 uv = uf[f], hv = hf[f];
    sbuf[f] = make_float2(uv.x * hv.x - uv.y * hv.y,
                          -(uv.x * hv.y + uv.y * hv.x));   // conj(U*H)
  }
  __syncthreads();
  fft_lds(sbuf, tid);                       // ifft(t) = Re(fft(conj(P)))/N
  float gg = g[ch], bc = bb[ch], mc = mu[ch], inv = rsqrtf(var[ch] + EPS_);
  _Float16* dst = mbn + ((size_t)b * M_ + ch) * S_;
  const float scale = 1.f / (float)NFFT;
  for (int t = tid; t < S_; t += 256) {
    float mval = sbuf[t].x * scale;
    dst[t] = (_Float16)(gg * (mval - mc) * inv + bc);
  }
}

// ---------------- LIF scan for m; write spikes channel-contiguous ----------------
__global__ void __launch_bounds__(256)
k_mlif(const _Float16* __restrict__ mbn, _Float16* __restrict__ ms) {
  int b = blockIdx.x, ch = threadIdx.x;     // 256 channels per block
  const _Float16* src = mbn + ((size_t)b * M_ + ch) * S_;
  float v = 0.f;
  for (int t = 0; t < S_; ++t) {
    float xv = (float)src[t];
    v = 0.5f * (v + xv);
    float s = (v >= 1.0f) ? 1.f : 0.f;
    ms[((size_t)b * S_ + t) * M_ + ch] = (_Float16)s;   // coalesced across ch
    v *= (1.f - s);
  }
}

#if HAVE_TDM
// Build a 2D-tile D# (see CDNA5 ISA) and issue
// TENSOR_LOAD_TO_LDS. lds_off = byte offset into workgroup LDS (smem at 0).
// tile: tile_d1 rows x tile_d0 elems of 2-byte data, row stride in elems.
__device__ __forceinline__ void tdm_load_2d(unsigned lds_off, const void* gptr,
                                            unsigned tile_d0, unsigned tile_d1,
                                            unsigned row_stride) {
  unsigned long long ga = (unsigned long long)(uintptr_t)gptr;
  v4u g0;
  g0[0] = 1u;                                               // count=1, user D#
  g0[1] = lds_off;                                          // lds_addr (bytes)
  g0[2] = (unsigned)(ga & 0xFFFFFFFFull);                   // global_addr[31:0]
  g0[3] = (unsigned)((ga >> 32) & 0x01FFFFFFull) | (2u << 30); // addr[56:32] | type=2
  v8i g1;
  unsigned td0 = row_stride;        // tensor dims >= tile dims (OOB guard)
  unsigned td1 = tile_d1;
  g1[0] = (int)(1u << 16);                                  // mask=0, data_size=1 (2B)
  g1[1] = (int)((td0 & 0xFFFFu) << 16);                     // tensor_dim0[15:0]
  g1[2] = (int)((td0 >> 16) | ((td1 & 0xFFFFu) << 16));     // dim0[31:16] | dim1[15:0]
  g1[3] = (int)((td1 >> 16) | (tile_d0 << 16));             // dim1[31:16] | tile_dim0
  g1[4] = (int)(tile_d1 & 0xFFFFu);                         // tile_dim1 (tile_dim2=0)
  g1[5] = (int)row_stride;                                  // tensor_dim0_stride[31:0]
  g1[6] = 0;                                                // stride[47:32] | dim1_stride lo
  g1[7] = 0;
  v4i gz = {0, 0, 0, 0};
#if TDM_6ARG
  v8i gz8 = {0, 0, 0, 0, 0, 0, 0, 0};
  __builtin_amdgcn_tensor_load_to_lds(g0, g1, gz, gz, gz8, 0);
#else
  __builtin_amdgcn_tensor_load_to_lds(g0, g1, gz, gz, 0);
#endif
}
#endif

// ---------------- WMMA GEMM: [65536 x 384] (concat(m_s,x)) @ [384 x 384] ----------------
// block tile 128(M) x 64(N), BK=32, 8 waves; each wave: 16 rows x 64 cols (4 wmma tiles)
// A/B tiles staged into double-buffered LDS by the Tensor Data Mover (TENSORcnt),
// overlapped with WMMA on the previous buffer.
__global__ void __launch_bounds__(256)
k_gemm(const _Float16* __restrict__ ms, const _Float16* __restrict__ xh,
       const _Float16* __restrict__ wh, float* __restrict__ out) {
  // [ A0 : 128x32 | B0 : 64x32 | A1 | B1 ]  (halfs); assumes smem at LDS offset 0
  __shared__ __align__(16) _Float16 smem[2 * (128 * 32 + 64 * 32)];   // 24 KB
  const int ABUF = 128 * 32;            // halfs per A buffer
  const int BBUF = 64 * 32;
  const int BUF  = ABUF + BBUF;         // 6144 halfs per stage
  int tid = threadIdx.x;
  int wv = tid >> 5, lane = tid & 31;
  int hh = lane >> 4, rr = lane & 15;   // lane half / index-in-half
  int r0 = blockIdx.x * 128;
  int n0 = blockIdx.y * 64;
  v8f acc[4] = {};
  const int NKT = K_ / 32;              // 12 K-tiles

#if HAVE_TDM
  // ---- issue first A/B tiles via TDM (one wave issues; EXEC-independent DMA)
  if (tid < 32) {
    const _Float16* asrc = ms + (size_t)r0 * M_;            // kk=0 < 256
    tdm_load_2d(0u, asrc, 32u, 128u, (unsigned)M_);
    const _Float16* bsrc = wh + (size_t)n0 * K_;
    tdm_load_2d((unsigned)(ABUF * 2), bsrc, 32u, 64u, (unsigned)K_);
  }
  __builtin_amdgcn_s_wait_tensorcnt(0);
  lds_written_by_dma(smem);
  __syncthreads();

  for (int kt = 0; kt < NKT; ++kt) {
    int cur = kt & 1;
    int kk = kt * 32;
    // prefetch next K-tile into the other buffer while we compute
    if (kt + 1 < NKT && tid < 32) {
      int nk = kk + 32;
      unsigned nbase = (unsigned)(((kt + 1) & 1) * BUF * 2);   // byte offset of stage
      const _Float16* asrc = (nk < M_)
          ? ms + (size_t)r0 * M_ + nk
          : xh + (size_t)r0 * I_ + (nk - M_);
      unsigned astr = (nk < M_) ? (unsigned)M_ : (unsigned)I_;
      tdm_load_2d(nbase, asrc, 32u, 128u, astr);
      tdm_load_2d(nbase + (unsigned)(ABUF * 2),
                  wh + (size_t)n0 * K_ + nk, 32u, 64u, (unsigned)K_);
    }
    const _Float16* As = smem + cur * BUF;
    const _Float16* Bs = smem + cur * BUF + ABUF;
#else
  for (int kt = 0; kt < NKT; ++kt) {
    int kk = kt * 32;
    _Float16* As = smem;                       // single buffer fallback
    _Float16* Bs = smem + ABUF;
    #pragma unroll
    for (int j = 0; j < 2; ++j) {
      int c = tid + j * 256;
      int row = c >> 2, cp = c & 3;
      const _Float16* src = (kk < M_)
          ? ms + ((size_t)(r0 + row)) * M_ + kk + cp * 8
          : xh + ((size_t)(r0 + row)) * I_ + (kk - M_) + cp * 8;
      *(H8*)&As[row * 32 + cp * 8] = *(const H8*)src;
    }
    {
      int row = tid >> 2, cp = tid & 3;
      *(H8*)&Bs[row * 32 + cp * 8] = *(const H8*)(wh + ((size_t)(n0 + row)) * K_ + kk + cp * 8);
    }
    __syncthreads();
#endif

    // A fragment (16x32 f16): lanes 0-15 rows w/ K 0-7,16-23; lanes 16-31 K 8-15,24-31
    v16h afrag;
    {
      int arow = wv * 16 + rr;
      H8 lo = *(const H8*)&As[arow * 32 + hh * 8];
      H8 hi = *(const H8*)&As[arow * 32 + 16 + hh * 8];
      #pragma unroll
      for (int i = 0; i < 8; ++i) { afrag[i] = lo.h[i]; afrag[i + 8] = hi.h[i]; }
    }
    #pragma unroll
    for (int nt = 0; nt < 4; ++nt) {
      v16h bfrag;                      // B (32x16): lane -> column n, K 0-7,16-23 / 8-15,24-31 by lane half
      int bn = nt * 16 + rr;
      H8 b0 = *(const H8*)&Bs[bn * 32 + hh * 8];
      H8 b1 = *(const H8*)&Bs[bn * 32 + 16 + hh * 8];
      #pragma unroll
      for (int i = 0; i < 8; ++i) { bfrag[i] = b0.h[i]; bfrag[i + 8] = b1.h[i]; }
      acc[nt] = __builtin_amdgcn_wmma_f32_16x16x32_f16(false, afrag, false, bfrag,
                                                       (short)0, acc[nt], false, false);
    }
#if HAVE_TDM
    __builtin_amdgcn_s_wait_tensorcnt(0);   // prefetch done (issuing wave); NOP for rest
    lds_written_by_dma(smem);
    __syncthreads();
#else
    __syncthreads();
#endif
  }

  // C/D layout: lanes 0-15 -> N=lane, VGPR v -> M=v (+8 for upper lanes)
  #pragma unroll
  for (int nt = 0; nt < 4; ++nt) {
    int col = n0 + nt * 16 + rr;
    #pragma unroll
    for (int v = 0; v < 8; ++v) {
      int row = r0 + wv * 16 + hh * 8 + v;
      out[(size_t)row * HD_ + col] = acc[nt][v];
    }
  }
}

// ---------------- bias + BN + LIF for h (in place on d_out), emit h_n ----------------
__global__ void __launch_bounds__(256)
k_hlif(float* __restrict__ h, const float* __restrict__ wb,
       const float* __restrict__ g, const float* __restrict__ bb,
       const float* __restrict__ mu, const float* __restrict__ var,
       float* __restrict__ hn) {
  int idx = blockIdx.x * blockDim.x + threadIdx.x;
  if (idx >= B_ * HD_) return;
  int b = idx / HD_, hd = idx - b * HD_;
  float bias = wb[hd], gg = g[hd], bc = bb[hd], mc = mu[hd], inv = rsqrtf(var[hd] + EPS_);
  float v = 0.f, s = 0.f;
  for (int t = 0; t < S_; ++t) {
    size_t o = ((size_t)b * S_ + t) * HD_ + hd;     // coalesced across hd
    float xbn = gg * (h[o] + bias - mc) * inv + bc;
    v = 0.5f * (v + xbn);
    s = (v >= 1.0f) ? 1.f : 0.f;
    h[o] = s;
    v *= (1.f - s);
  }
  hn[idx] = s;                                      // h[:, -1, :]
}

// ---------------- launcher ----------------
extern "C" void kernel_launch(void* const* d_in, const int* in_sizes, int n_in,
                              void* d_out, int out_size, void* d_ws, size_t ws_size,
                              hipStream_t stream) {
  (void)in_sizes; (void)n_in; (void)out_size; (void)ws_size;
  const float* x     = (const float*)d_in[0];
  const float* Hm    = (const float*)d_in[1];
  const float* Wuw   = (const float*)d_in[2];
  const float* Wub   = (const float*)d_in[3];
  const float* Whw   = (const float*)d_in[4];
  const float* Whb   = (const float*)d_in[5];
  const float* g_u   = (const float*)d_in[6];
  const float* b_u   = (const float*)d_in[7];
  const float* mu_u  = (const float*)d_in[8];
  const float* var_u = (const float*)d_in[9];
  const float* g_m   = (const float*)d_in[10];
  const float* b_m   = (const float*)d_in[11];
  const float* mu_m  = (const float*)d_in[12];
  const float* var_m = (const float*)d_in[13];
  const float* g_h   = (const float*)d_in[14];
  const float* b_h   = (const float*)d_in[15];
  const float* mu_h  = (const float*)d_in[16];
  const float* var_h = (const float*)d_in[17];

  char* ws = (char*)d_ws;
  size_t off = 0;
  auto carve = [&](size_t bytes) -> char* {
    char* p = ws + off;
    off = (off + bytes + 255) & ~(size_t)255;
    return p;
  };
  float*    u_buf = (float*)   carve((size_t)B_ * S_ * 4);        // u_bn -> u spikes
  float2*   Uf    = (float2*)  carve((size_t)B_ * NFFT * 8);      // 1 MB
  float2*   Hf    = (float2*)  carve((size_t)M_ * NFFT * 8);      // 16 MB
  _Float16* x_h   = (_Float16*)carve((size_t)B_ * S_ * I_ * 2);   // 16.8 MB
  _Float16* wh    = (_Float16*)carve((size_t)HD_ * K_ * 2);       // 0.3 MB
  _Float16* m_bn  = (_Float16*)carve((size_t)B_ * M_ * S_ * 2);   // 33.5 MB
  _Float16* m_s   = (_Float16*)carve((size_t)B_ * S_ * M_ * 2);   // 33.5 MB

  float* hout = (float*)d_out;                       // h [B,S,HD]
  float* hn   = hout + (size_t)B_ * S_ * HD_;        // h_n [B,HD]

  k_cvt_f2h<<<(B_ * S_ * I_ + 255) / 256, 256, 0, stream>>>(x, x_h, B_ * S_ * I_);
  k_cvt_f2h<<<(HD_ * K_ + 255) / 256, 256, 0, stream>>>(Whw, wh, HD_ * K_);
  k_upre<<<(B_ * S_) / 8, 256, 0, stream>>>(x, Wuw, Wub, g_u, b_u, mu_u, var_u, u_buf);
  k_ulif<<<1, 32, 0, stream>>>(u_buf);
  k_fft_fwd<<<B_, 256, 0, stream>>>(u_buf, Uf, S_);
  k_fft_fwd<<<M_, 256, 0, stream>>>(Hm, Hf, S_);
  k_conv_ifft<<<B_ * M_, 256, 0, stream>>>(Uf, Hf, g_m, b_m, mu_m, var_m, m_bn);
  k_mlif<<<B_, 256, 0, stream>>>(m_bn, m_s);
  dim3 ggrid((B_ * S_) / 128, HD_ / 64);             // (512, 6)
  k_gemm<<<ggrid, 256, 0, stream>>>(m_s, x_h, wh, hout);
  k_hlif<<<(B_ * HD_ + 255) / 256, 256, 0, stream>>>(hout, Whb, g_h, b_h, mu_h, var_h, hn);
}
